// MultiHead_Attention_Lattice_rel_save_gpumm_25580825215160
// MI455X (gfx1250) — matrix operand order repaired and checked by reading, not compiled, not run on hardware
//
#include <hip/hip_runtime.h>
#include <hip/hip_bf16.h>

#define B_  2
#define L_  256
#define H_  512
#define NH_ 8
#define DH_ 64

typedef __attribute__((ext_vector_type(16))) __bf16 v16bf;
typedef __attribute__((ext_vector_type(8)))  float  v8f;

union Frag { v16bf v; unsigned int u[8]; };

// pack two f32 -> packed bf16 (single v_perm_b32: bytes {hi.3,hi.2,lo.3,lo.2})
__device__ __forceinline__ unsigned int pack2bf(float lo, float hi) {
#if __has_builtin(__builtin_amdgcn_perm)
  return __builtin_amdgcn_perm(__float_as_uint(hi), __float_as_uint(lo),
                               0x07060302u);
#else
  unsigned int a = __float_as_uint(lo) >> 16;
  unsigned int b = __float_as_uint(hi) & 0xFFFF0000u;
  return b | a;
#endif
}

// load two contiguous f32 (8B aligned) and pack to bf16x2
__device__ __forceinline__ unsigned int load_pack(const float* p) {
  const float2 t = *(const float2*)p;
  return pack2bf(t.x, t.y);
}

// O = X @ W^T + bias.  X:[512,512] row-major, W:[512,512] row-major (torch Linear weight)
// mode 0: store O as [B,NH,L,DH] head-major; mode 1: store O as [M,H] row-major.
__global__ void gemm_xt_kernel(const float* __restrict__ X,
                               const float* __restrict__ W,
                               const float* __restrict__ bias,
                               float* __restrict__ O, int mode) {
  const int ln = threadIdx.x, l16 = ln & 15, hh = ln >> 4;
  const int row = blockIdx.x * 16 + l16;      // A-fragment row (M)
  const int col = blockIdx.y * 16 + l16;      // B-fragment col (N) and D col
  const int kbA = hh * 8, kbB = hh * 16;
  v8f c = {};
  for (int k0 = 0; k0 < H_; k0 += 32) {
    if (k0 + 32 < H_) {
      __builtin_prefetch(X + row * H_ + k0 + 32, 0, 1);   // global_prefetch_b8
      __builtin_prefetch(W + col * H_ + k0 + 32, 0, 1);
    }
    Frag a, b;
#pragma unroll
    for (int j = 0; j < 8; ++j) {
      const int ka = k0 + kbB + (j << 1);
      a.u[j] = load_pack(X + row * H_ + ka);
      const int kb = k0 + kbB + (j << 1);
      b.u[j] = load_pack(W + col * H_ + kb);
    }
    c = __builtin_amdgcn_wmma_f32_16x16x32_bf16(false, a.v, false, b.v,
                                                (short)0, c, false, false);
  }
  const float bv = bias[col];
#pragma unroll
  for (int j = 0; j < 8; ++j) {
    const int m = blockIdx.x * 16 + j + hh * 8;
    const float val = c[j] + bv;
    if (mode == 0) {
      const int bb = m >> 8, l = m & 255;
      const int n = col >> 6, d = col & 63;
      O[(((bb * NH_ + n) * L_) + l) * DH_ + d] = val;
    } else {
      O[m * H_ + col] = val;
    }
  }
}

// WQ[b,q,n,h] = sum_d (Qp[b,n,q,d] + vbias[n,d]) * Wr[n*64+d, h]
__global__ void wq_kernel(const float* __restrict__ Qp,
                          const float* __restrict__ Wr,
                          const float* __restrict__ vbias,
                          float* __restrict__ WQ) {
  const int ln = threadIdx.x, l16 = ln & 15, hh = ln >> 4;
  const int bn = blockIdx.z, b = bn >> 3, n = bn & 7;
  const int qrow = blockIdx.x * 16 + l16;
  const int col  = blockIdx.y * 16 + l16;
  const int kbA = hh * 8, kbB = hh * 16;
  const float* Qb = Qp + ((b * NH_ + n) * L_) * DH_;
  v8f c = {};
#pragma unroll
  for (int k0 = 0; k0 < DH_; k0 += 32) {
    Frag a, bb;
#pragma unroll
    for (int j = 0; j < 8; ++j) {
      const int d = k0 + kbB + (j << 1);
      const float2 qa = *(const float2*)(Qb + qrow * DH_ + d);
      const float2 vb = *(const float2*)(vbias + n * DH_ + d);
      a.u[j] = pack2bf(qa.x + vb.x, qa.y + vb.y);
      const int kd = k0 + kbB + (j << 1);
      bb.u[j] = pack2bf(Wr[(n * DH_ + kd) * H_ + col],
                        Wr[(n * DH_ + kd + 1) * H_ + col]);
    }
    c = __builtin_amdgcn_wmma_f32_16x16x32_bf16(false, a.v, false, bb.v,
                                                (short)0, c, false, false);
  }
#pragma unroll
  for (int j = 0; j < 8; ++j) {
    const int q = blockIdx.x * 16 + j + hh * 8;
    WQ[((b * L_ + q) * NH_ + n) * H_ + col] = c[j];
  }
}

// brdot[b,n,q] = sum_d (Qp + vbias) * br
__global__ void brdot_kernel(const float* __restrict__ Qp,
                             const float* __restrict__ vbias,
                             const float* __restrict__ br,
                             float* __restrict__ BRD) {
  const int t = blockIdx.x * blockDim.x + threadIdx.x;
  if (t >= B_ * NH_ * L_) return;
  const int q = t & (L_ - 1), bn = t >> 8, n = bn & 7;
  const float* Qb = Qp + (bn * L_ + q) * DH_;
  float s = 0.f;
  for (int d = 0; d < DH_; ++d)
    s += (Qb[d] + vbias[n * DH_ + d]) * br[n * DH_ + d];
  BRD[t] = s;
}

// issue one 16x32-float R chunk (2 KB) into LDS via 4 wave-wide async b128 copies
__device__ __forceinline__ void async_chunk(unsigned ldsb, const float* grow_k0,
                                            int l16, int hh) {
  const unsigned base = ldsb + (unsigned)(l16 * 128 + hh * 16);
  const float* g = grow_k0 + hh * 4;
#pragma unroll
  for (int i = 0; i < 4; ++i) {
    asm volatile("global_load_async_to_lds_b128 %0, %1, off"
                 :: "v"(base + (unsigned)(i * 32)), "v"(g + i * 8)
                 : "memory");
  }
}

// SB[b,n,q,k] = sum_h WQ[b,q,n,h] * R[b,q,k,h]   (streams rel_pos_embedding once,
// double-buffered through LDS with async global->LDS copies, ASYNCcnt-tracked)
__global__ void bd_kernel(const float* __restrict__ WQ,
                          const float* __restrict__ R,
                          float* __restrict__ SB) {
  __shared__ float rbuf[2][16 * 32];       // 2 x 2KB chunk buffers
  const int ln = threadIdx.x, l16 = ln & 15, hh = ln >> 4;
  const int bq = blockIdx.x;               // b*L + q
  const int b  = bq >> 8, q = bq & 255;
  const int kcol = blockIdx.y * 16 + l16;
  const int kbA = hh * 8, kbB = hh * 16;
  const float* Arow = WQ + ((long)bq * NH_ + (l16 & 7)) * H_;  // M rows 0..7 real
  const bool arow_valid = (l16 < 8);
  // this lane's R row for the async copies (row index within tile = l16)
  const float* Rrow = R + ((long)bq * L_ + blockIdx.y * 16 + l16) * H_;
  const unsigned lds0 = (unsigned)(size_t)&rbuf[0][0];
  const unsigned lds1 = (unsigned)(size_t)&rbuf[1][0];

  async_chunk(lds0, Rrow, l16, hh);        // prefill chunk 0

  v8f c = {};
  for (int k0 = 0; k0 < H_; k0 += 32) {
    const unsigned nxt = ((k0 >> 5) & 1) ? lds0 : lds1;
    if (k0 + 32 < H_) {
      async_chunk(nxt, Rrow + k0 + 32, l16, hh);
      asm volatile("s_wait_asynccnt 0x4" ::: "memory");  // chunk k0 landed
    } else {
      asm volatile("s_wait_asynccnt 0x0" ::: "memory");
    }
    const float* cbuf = (const float*)(&rbuf[((k0 >> 5) & 1)][0]);
    Frag a, bb;
#pragma unroll
    for (int j = 0; j < 8; ++j) {
      const int ka = k0 + kbB + (j << 1);
      const float2 av = *(const float2*)(Arow + ka);
      a.u[j] = pack2bf(arow_valid ? av.x : 0.f, arow_valid ? av.y : 0.f);
      // B operand from the LDS chunk: row = l16, cols kbB+2j within chunk
      bb.u[j] = load_pack(cbuf + l16 * 32 + kbB + (j << 1));
    }
    c = __builtin_amdgcn_wmma_f32_16x16x32_bf16(false, a.v, false, bb.v,
                                                (short)0, c, false, false);
  }
  if (ln < 16) {                           // D rows 0..7 (= heads) live in lanes 0-15
#pragma unroll
    for (int j = 0; j < 8; ++j)
      SB[(((long)(b * NH_ + j) * L_ + q) * L_) + kcol] = c[j];
  }
}

// scores = (Qu@K^T + SB + brdot)*0.125 -> masked softmax -> P@V -> O1[B,L,H]
__global__ void attn_kernel(const float* __restrict__ Qp,
                            const float* __restrict__ Kp,
                            const float* __restrict__ Vp,
                            const float* __restrict__ SB,
                            const float* __restrict__ BRD,
                            const float* __restrict__ ubias,
                            const int* __restrict__ seq_len,
                            const int* __restrict__ lex_num,
                            float* __restrict__ O1) {
  __shared__ float lds[16][L_ + 4];
  const int ln = threadIdx.x, l16 = ln & 15, hh = ln >> 4;
  const int bn = blockIdx.x, b = bn >> 3, n = bn & 7;
  const int qt = blockIdx.y;
  const int kbA = hh * 8, kbB = hh * 16;
  const int total = seq_len[b] + lex_num[b];
  const float* Qb = Qp + (long)(bn * L_) * DH_;
  const float* Kb = Kp + (long)(bn * L_) * DH_;
  const float* Vb = Vp + (long)(bn * L_) * DH_;

  const int qrow = qt * 16 + l16;
  float bd[8];
#pragma unroll
  for (int j = 0; j < 8; ++j)
    bd[j] = BRD[bn * L_ + qt * 16 + j + hh * 8];

  // Qu fragments for both K-steps (DH=64)
  Frag afr[2];
#pragma unroll
  for (int s = 0; s < 2; ++s)
#pragma unroll
    for (int j = 0; j < 8; ++j) {
      const int d = s * 32 + kbB + (j << 1);
      const float2 qa = *(const float2*)(Qb + qrow * DH_ + d);
      const float2 ub = *(const float2*)(ubias + n * DH_ + d);
      afr[s].u[j] = pack2bf(qa.x + ub.x, qa.y + ub.y);
    }

  v8f st[16];
  for (int t = 0; t < 16; ++t) {
    const int kg = t * 16 + l16;
    v8f c;
#pragma unroll
    for (int j = 0; j < 8; ++j)      // init accumulator from precomputed B_D scores
      c[j] = SB[((long)(bn * L_) + qt * 16 + j + hh * 8) * L_ + kg];
#pragma unroll
    for (int s = 0; s < 2; ++s) {
      Frag bb;
#pragma unroll
      for (int j = 0; j < 8; ++j) {
        const int kd = s * 32 + kbB + (j << 1);
        bb.u[j] = load_pack(Kb + kg * DH_ + kd);
      }
      c = __builtin_amdgcn_wmma_f32_16x16x32_bf16(false, afr[s].v, false, bb.v,
                                                  (short)0, c, false, false);
    }
    st[t] = c;
  }

  // scale + brdot + mask, rowwise max
  float mx[8];
#pragma unroll
  for (int j = 0; j < 8; ++j) mx[j] = -3.0e38f;
  for (int t = 0; t < 16; ++t) {
    const int kg = t * 16 + l16;
    const bool live = kg < total;
#pragma unroll
    for (int j = 0; j < 8; ++j) {
      float v = (st[t][j] + bd[j]) * 0.125f;
      v = live ? v : -1.0e30f;
      st[t][j] = v;
      mx[j] = fmaxf(mx[j], v);
    }
  }
#pragma unroll
  for (int j = 0; j < 8; ++j)
    for (int m = 1; m < 16; m <<= 1)
      mx[j] = fmaxf(mx[j], __shfl_xor(mx[j], m, 32));

  float sm[8];
#pragma unroll
  for (int j = 0; j < 8; ++j) sm[j] = 0.f;
  for (int t = 0; t < 16; ++t)
#pragma unroll
    for (int j = 0; j < 8; ++j) {
      const float e = __expf(st[t][j] - mx[j]);
      st[t][j] = e;
      sm[j] += e;
    }
#pragma unroll
  for (int j = 0; j < 8; ++j) {
    for (int m = 1; m < 16; m <<= 1)
      sm[j] += __shfl_xor(sm[j], m, 32);
    sm[j] = 1.0f / sm[j];
  }

  // transpose probabilities D-layout -> A-layout via LDS
  for (int t = 0; t < 16; ++t)
#pragma unroll
    for (int j = 0; j < 8; ++j)
      lds[j + hh * 8][t * 16 + l16] = st[t][j];
  __syncthreads();

  // O = P @ V  (M=16 q, N=64 d, K=256 k)
#pragma unroll
  for (int dt = 0; dt < 4; ++dt) {
    const int dcol = dt * 16 + l16;
    v8f acc = {};
    for (int k0 = 0; k0 < L_; k0 += 32) {
      Frag a, bb;
#pragma unroll
      for (int j = 0; j < 8; ++j) {
        const int kk = k0 + kbB + (j << 1);
        a.u[j] = load_pack(&lds[l16][kk]);
        const int kr = k0 + kbB + (j << 1);
        bb.u[j] = pack2bf(Vb[kr * DH_ + dcol], Vb[(kr + 1) * DH_ + dcol]);
      }
      acc = __builtin_amdgcn_wmma_f32_16x16x32_bf16(false, a.v, false, bb.v,
                                                    (short)0, acc, false, false);
    }
#pragma unroll
    for (int j = 0; j < 8; ++j) {
      const int q = qt * 16 + j + hh * 8;
      O1[(long)(b * L_ + q) * H_ + n * DH_ + dcol] = acc[j] * sm[j];
    }
  }
}

extern "C" void kernel_launch(void* const* d_in, const int* in_sizes, int n_in,
                              void* d_out, int out_size, void* d_ws, size_t ws_size,
                              hipStream_t stream) {
  const float* key   = (const float*)d_in[0];
  const float* query = (const float*)d_in[1];
  const float* value = (const float*)d_in[2];
  const float* rel   = (const float*)d_in[3];
  const float* Wk = (const float*)d_in[4];
  const float* bk = (const float*)d_in[5];
  const float* Wq = (const float*)d_in[6];
  const float* bq = (const float*)d_in[7];
  const float* Wv = (const float*)d_in[8];
  const float* bv = (const float*)d_in[9];
  const float* Wr = (const float*)d_in[10];
  const float* br = (const float*)d_in[11];
  const float* ub = (const float*)d_in[12];
  const float* vb = (const float*)d_in[13];
  const float* Wf = (const float*)d_in[14];
  const float* bfv = (const float*)d_in[15];
  const int* seq_len = (const int*)d_in[16];
  const int* lex_num = (const int*)d_in[17];
  float* out = (float*)d_out;

  // workspace carve-up (floats)
  float* ws  = (float*)d_ws;
  float* Qp  = ws;                         // [B,NH,L,DH]  262144
  float* Kp  = Qp  + 262144;               // [B,NH,L,DH]  262144
  float* Vp  = Kp  + 262144;               // [B,NH,L,DH]  262144
  float* WQ  = Vp  + 262144;               // [B,L,NH,H]   2097152
  float* SB  = WQ  + 2097152;              // [B,NH,L,L]   4194304
  float* BRD = SB  + 4194304;              // [B,NH,L]     4096
  float* O1  = BRD + 4096;                 // [B,L,H]      262144

  const dim3 blk(32);
  const dim3 gproj(32, 32);
  gemm_xt_kernel<<<gproj, blk, 0, stream>>>(key,   Wk, bk, Kp, 0);
  gemm_xt_kernel<<<gproj, blk, 0, stream>>>(query, Wq, bq, Qp, 0);
  gemm_xt_kernel<<<gproj, blk, 0, stream>>>(value, Wv, bv, Vp, 0);

  wq_kernel<<<dim3(16, 32, 16), blk, 0, stream>>>(Qp, Wr, vb, WQ);
  brdot_kernel<<<16, 256, 0, stream>>>(Qp, vb, br, BRD);

  bd_kernel<<<dim3(512, 16), blk, 0, stream>>>(WQ, rel, SB);

  attn_kernel<<<dim3(16, 16), blk, 0, stream>>>(Qp, Kp, Vp, SB, BRD, ub,
                                                seq_len, lex_num, O1);

  gemm_xt_kernel<<<gproj, blk, 0, stream>>>(O1, Wf, bfv, out, 1);
}
